// PointerDecoder_24799141167548
// MI455X (gfx1250) — matrix operand order repaired and checked by reading, not compiled, not run on hardware
//
#include <hip/hip_runtime.h>

// ---------------------------------------------------------------------------
// Pointer-network decoder for MI455X (gfx1250, wave32, WMMA bf16 -> f32)
// ---------------------------------------------------------------------------
#define BB 32
#define TT 64
#define NN 2048
#define SS 2048
#define DD 256   // DEMB = DENC = DDEC = 256
#define G4 1024  // 4*H

typedef __attribute__((ext_vector_type(16))) __bf16 v16bf;
typedef __attribute__((ext_vector_type(8)))  float  v8f;
typedef __attribute__((ext_vector_type(4)))  unsigned int u32x4;

// ---- WMMA wrapper ----------------------------------------------------------
__device__ inline v8f wmma_bf16(v16bf a, v16bf b, v8f c) {
  return __builtin_amdgcn_wmma_f32_16x16x32_bf16(
      /*neg_a=*/false, a, /*neg_b=*/false, b,
      /*c_mod=*/(short)0, c, /*reuse_a=*/false, /*reuse_b=*/false);
}

// ---- Fragment loaders (bf16 A, B: contiguous k = 16h+s, not ISA table order) --------------------------------
// A-matrix 16x32 (MxK) bf16: lane<16 -> K 0..15 ; lane>=16 -> K 16..31
__device__ inline v16bf fragA_bf16(const __bf16* p, int ld) {
  const int lane = threadIdx.x & 31;
  const int r = lane & 15;
  const int hi = (lane >> 4) << 4;            // 0 or 16
  const __bf16* q = p + r * ld + hi;
  union { v16bf v; u32x4 u[2]; } t;
  t.u[0] = *(const u32x4*)(q);
  t.u[1] = *(const u32x4*)(q + 8);
  return t.v;
}
__device__ inline v16bf fragA_f32(const float* p, int ld) {
  const int lane = threadIdx.x & 31;
  const int r = lane & 15;
  const int hi = (lane >> 4) << 3;
  const float* q = p + r * ld + hi;
  v16bf v;
#pragma unroll
  for (int i = 0; i < 8; ++i) v[i] = (__bf16)q[i];
#pragma unroll
  for (int i = 0; i < 8; ++i) v[8 + i] = (__bf16)q[16 + i];
  return v;
}
// B-matrix 32x16 (KxN) stored as N-major rows (Bt[n][k]): lane<16 -> K 0..15
__device__ inline v16bf fragB_bf16(const __bf16* p, int ld) {
  const int lane = threadIdx.x & 31;
  const int n = lane & 15;
  const int kh = (lane >> 4) << 4;            // 0 or 16
  const __bf16* q = p + n * ld + kh;
  union { v16bf v; u32x4 u[2]; } t;
  t.u[0] = *(const u32x4*)(q);
  t.u[1] = *(const u32x4*)(q + 8);
  return t.v;
}
__device__ inline v16bf fragB_f32(const float* p, int ld) {
  const int lane = threadIdx.x & 31;
  const int n = lane & 15;
  const int kh = (lane >> 4) << 4;
  const float* q = p + n * ld + kh;
  v16bf v;
#pragma unroll
  for (int i = 0; i < 16; ++i) v[i] = (__bf16)q[i];
  return v;
}
// C/D 16x16 f32: element r of lane L is (M = r + 8*(L>=16), N = L&15)
__device__ inline void store_tile_f32(float* p, int ld, v8f d) {
  const int lane = threadIdx.x & 31;
  const int n = lane & 15;
  const int m0 = (lane >> 4) << 3;
#pragma unroll
  for (int r = 0; r < 8; ++r) p[(size_t)(m0 + r) * ld + n] = d[r];
}
__device__ inline void store_tile_bf16(__bf16* p, int ld, v8f d) {
  const int lane = threadIdx.x & 31;
  const int n = lane & 15;
  const int m0 = (lane >> 4) << 3;
#pragma unroll
  for (int r = 0; r < 8; ++r) p[(size_t)(m0 + r) * ld + n] = (__bf16)d[r];
}

__device__ inline float sigmoidf(float x) { return 1.f / (1.f + __expf(-x)); }

__device__ inline v8f v8f_zero() {
  v8f z;
#pragma unroll
  for (int r = 0; r < 8; ++r) z[r] = 0.f;
  return z;
}

// ---------------------------------------------------------------------------
// Prep: bf16 weight packing + firstpick init
// ---------------------------------------------------------------------------
__global__ void prep_kernel(const float* __restrict__ Wih, const float* __restrict__ Whh,
                            const float* __restrict__ bih, const float* __restrict__ bhh,
                            const float* __restrict__ Wdt, const float* __restrict__ Wf,
                            __bf16* __restrict__ Wcat, float* __restrict__ biascat,
                            __bf16* __restrict__ Wdtb, __bf16* __restrict__ Wfb,
                            int* __restrict__ firstpick) {
  const int stride = gridDim.x * blockDim.x;
  const int t0 = blockIdx.x * blockDim.x + threadIdx.x;
  for (int i = t0; i < G4 * 512; i += stride) {          // Wcat[col][k], k = [x ; h]
    int col = i >> 9, k = i & 511;
    float v = (k < 256) ? Wih[col * 256 + k] : Whh[col * 256 + (k - 256)];
    Wcat[i] = (__bf16)v;
  }
  for (int i = t0; i < G4; i += stride) biascat[i] = bih[i] + bhh[i];
  for (int i = t0; i < DD * DD; i += stride) Wdtb[i] = (__bf16)Wdt[i];
  for (int i = t0; i < 272 * 512; i += stride) {         // Wf padded to 272 rows
    int row = i >> 9, k = i & 511;
    Wfb[i] = (__bf16)((row < 257) ? Wf[row * 512 + k] : 0.f);
  }
  for (int i = t0; i < BB * NN; i += stride) firstpick[i] = 1 << 30;
}

__global__ void fp_fill_kernel(const int* __restrict__ gt, int* __restrict__ firstpick) {
  int i = blockIdx.x * blockDim.x + threadIdx.x;
  if (i < BB * TT) {
    int b = i >> 6, t = i & 63;
    atomicMin(&firstpick[b * NN + gt[i]], t);
  }
}

// LSTM inputs X[t][b][d]: go at t=0, else emb[b][gt[b][t-1]][:]
__global__ void build_x_kernel(const float* __restrict__ emb, const int* __restrict__ gt,
                               const float* __restrict__ go, __bf16* __restrict__ X) {
  int t = blockIdx.x >> 5;
  int b = blockIdx.x & 31;
  int d = threadIdx.x;
  float v = (t == 0) ? go[d]
                     : emb[((size_t)b * NN + gt[b * TT + t - 1]) * DD + d];
  X[(size_t)blockIdx.x * DD + d] = (__bf16)v;
}

// ---------------------------------------------------------------------------
// LSTM: one 1024-thread workgroup, 64 serial steps, gates via WMMA.
// Wave w owns column tiles {ntb, ntb+16, ntb+32, ntb+48} = same hidden slice
// of i/f/g/o  ->  gate fusion + cell state live entirely in registers.
// A laundered VGPR zero-offset makes the weight addresses formally
// loop-variant so the 64 B-fragments are re-streamed from L2 each step
// (global_load_b128) instead of being hoisted and spilled to scratch.
// ---------------------------------------------------------------------------
__global__ __launch_bounds__(1024) void lstm_kernel(const __bf16* __restrict__ X,
                                                    const __bf16* __restrict__ Wcat,
                                                    const float* __restrict__ biascat,
                                                    __bf16* __restrict__ hs) {
  __shared__ __bf16 act[BB][512];                 // [b][x(256) | h(256)]
  const int tid = threadIdx.x;
  const int lane = tid & 31;
  const int w = tid >> 5;                         // wave 0..31
  const int mt = w >> 4;                          // 0..1  (row tile)
  const int ntb = w & 15;                         // 0..15 (hidden tile)
  const int n = lane & 15;
  const int mo = (lane >> 4) << 3;                // 0 or 8
  const int j = ntb * 16 + n;                     // hidden unit 0..255

  float c8[8];
#pragma unroll
  for (int r = 0; r < 8; ++r) c8[r] = 0.f;
  for (int e = tid; e < BB * 256; e += 1024)      // h0 = 0
    act[e >> 8][256 + (e & 255)] = (__bf16)0.f;

  const float bi = biascat[j];
  const float bfg = biascat[256 + j];
  const float bg = biascat[512 + j];
  const float bo = biascat[768 + j];

  // Wave-private weight base (column block ntb of all four gates).
  const __bf16* Wbase = Wcat + (size_t)(ntb * 16) * 512;

  for (int t = 0; t < TT; ++t) {
    // Defeat LICM: laundered zero (lives in a VGPR) makes the weight
    // addresses loop-variant each iteration.
    unsigned lz = 0;
    asm volatile("" : "+v"(lz));
    const __bf16* Wt = Wbase + lz;

    for (int e = tid; e < BB * 256; e += 1024)    // load x_t
      act[e >> 8][e & 255] = X[(size_t)t * BB * 256 + e];
    __syncthreads();

    v8f acc[4];
#pragma unroll
    for (int g = 0; g < 4; ++g) acc[g] = v8f_zero();
#pragma unroll 1
    for (int kt = 0; kt < 16; ++kt) {
      v16bf a = fragA_bf16(&act[mt * 16][kt * 32], 512);
#pragma unroll
      for (int g = 0; g < 4; ++g) {
        const __bf16* bp = Wt + (size_t)(g * 256) * 512 + kt * 32;
        acc[g] = wmma_bf16(a, fragB_bf16(bp, 512), acc[g]);
      }
    }
    __syncthreads();                              // all act reads done

#pragma unroll
    for (int r = 0; r < 8; ++r) {                 // i,f,g,o fused in registers
      const int m = mt * 16 + mo + r;             // batch row
      float iv = sigmoidf(acc[0][r] + bi);
      float fv = sigmoidf(acc[1][r] + bfg);
      float gv = tanhf(acc[2][r] + bg);
      float ov = sigmoidf(acc[3][r] + bo);
      c8[r] = fv * c8[r] + iv * gv;
      float h = ov * tanhf(c8[r]);
      __bf16 hb = (__bf16)h;
      act[m][256 + j] = hb;
      hs[((size_t)m * TT + t) * DD + j] = hb;
    }
    __syncthreads();
  }
}

// ---------------------------------------------------------------------------
// dec = hs @ Wdt^T + bdt   (M=2048, N=256, K=256) -> bf16
// ---------------------------------------------------------------------------
__global__ __launch_bounds__(256) void gemm_dec(const __bf16* __restrict__ A,
                                                const __bf16* __restrict__ Bw,
                                                const float* __restrict__ bias,
                                                __bf16* __restrict__ C) {
  const int wid = (blockIdx.x * 256 + threadIdx.x) >> 5;  // 2048 tiles
  const int mt = wid >> 4, nt = wid & 15;
  v8f acc = v8f_zero();
  for (int kt = 0; kt < 8; ++kt)
    acc = wmma_bf16(fragA_bf16(A + (size_t)(mt * 16) * DD + kt * 32, DD),
                    fragB_bf16(Bw + (size_t)(nt * 16) * DD + kt * 32, DD), acc);
  const float bv = bias[nt * 16 + (threadIdx.x & 15)];
#pragma unroll
  for (int r = 0; r < 8; ++r) acc[r] += bv;
  store_tile_bf16(C + (size_t)(mt * 16) * DD + nt * 16, DD, acc);
}

// ---------------------------------------------------------------------------
// raw[b][t][s] = dec[b] @ enc[b]^T  - (1-enc_mask)*1e20     (f32 out)
// enc converted f32->bf16 inside the B fragment loader.
// ---------------------------------------------------------------------------
__global__ __launch_bounds__(256) void gemm_raw(const __bf16* __restrict__ dec,
                                                const float* __restrict__ enc,
                                                const float* __restrict__ enc_mask,
                                                float* __restrict__ raw) {
  const int b = blockIdx.y;
  const int wid = (blockIdx.x * 256 + threadIdx.x) >> 5;  // 512 tiles/batch
  const int mt = wid >> 7, nt = wid & 127;
  const __bf16* A = dec + (size_t)b * TT * DD;
  const float* Bp = enc + (size_t)b * SS * DD;
  v8f acc = v8f_zero();
  for (int kt = 0; kt < 8; ++kt)
    acc = wmma_bf16(fragA_bf16(A + (size_t)(mt * 16) * DD + kt * 32, DD),
                    fragB_f32(Bp + (size_t)(nt * 16) * DD + kt * 32, DD), acc);
  const int s = nt * 16 + (threadIdx.x & 15);
  const float msk = (1.f - enc_mask[(size_t)b * SS + s]) * 1e20f;
#pragma unroll
  for (int r = 0; r < 8; ++r) acc[r] -= msk;
  store_tile_f32(raw + ((size_t)b * TT + mt * 16) * SS + nt * 16, SS, acc);
}

// ---------------------------------------------------------------------------
// softmax over s, write bf16 weights
// ---------------------------------------------------------------------------
__global__ __launch_bounds__(256) void softmax_kernel(const float* __restrict__ raw,
                                                      __bf16* __restrict__ wout) {
  __shared__ float red[256];
  const int row = blockIdx.x;
  const int tid = threadIdx.x;
  const float* p = raw + (size_t)row * SS;
  float ex[8], mx = -3.4e38f;
#pragma unroll
  for (int i = 0; i < 8; ++i) { ex[i] = p[tid + i * 256]; mx = fmaxf(mx, ex[i]); }
  red[tid] = mx; __syncthreads();
  for (int s = 128; s; s >>= 1) { if (tid < s) red[tid] = fmaxf(red[tid], red[tid + s]); __syncthreads(); }
  mx = red[0]; __syncthreads();
  float sum = 0.f;
#pragma unroll
  for (int i = 0; i < 8; ++i) { ex[i] = __expf(ex[i] - mx); sum += ex[i]; }
  red[tid] = sum; __syncthreads();
  for (int s = 128; s; s >>= 1) { if (tid < s) red[tid] += red[tid + s]; __syncthreads(); }
  const float inv = 1.f / red[0];
#pragma unroll
  for (int i = 0; i < 8; ++i) wout[(size_t)row * SS + tid + i * 256] = (__bf16)(ex[i] * inv);
}

// ---------------------------------------------------------------------------
// attn = w @ enc  (K = S = 2048). enc needed K-major -> transpose f32->bf16
// tiles through LDS per K-step.
// ---------------------------------------------------------------------------
__global__ __launch_bounds__(256) void gemm_attn(const __bf16* __restrict__ wsm,
                                                 const float* __restrict__ enc,
                                                 __bf16* __restrict__ attn) {
  __shared__ __bf16 Bt[256][40];                  // [d][s-in-tile], padded
  const int b = blockIdx.y, mt = blockIdx.x;      // mt: 0..3 (16 t-rows)
  const int tid = threadIdx.x, w = tid >> 5;
  const __bf16* A = wsm + ((size_t)b * TT + mt * 16) * SS;
  const float* E = enc + (size_t)b * SS * DD;
  v8f acc[2] = {v8f_zero(), v8f_zero()};
  for (int ks = 0; ks < SS / 32; ++ks) {
    for (int e = tid; e < 32 * 256; e += 256) {   // stage + transpose + cvt
      int si = e >> 8, d = e & 255;
      Bt[d][si] = (__bf16)E[(size_t)(ks * 32 + si) * DD + d];
    }
    __syncthreads();
    v16bf a = fragA_bf16(A + ks * 32, SS);
#pragma unroll
    for (int i = 0; i < 2; ++i)
      acc[i] = wmma_bf16(a, fragB_bf16(&Bt[(w * 2 + i) * 16][0], 40), acc[i]);
    __syncthreads();
  }
#pragma unroll
  for (int i = 0; i < 2; ++i)
    store_tile_bf16(attn + ((size_t)b * TT + mt * 16) * DD + (w * 2 + i) * 16, DD, acc[i]);
}

// ---------------------------------------------------------------------------
// feat = [hs | attn] @ Wf^T + bf   (M=2048, N=257 (pad 272), K=512)
// cols 0..255 -> bf16 feat ; col 256 -> f32 "picked" scale
// ---------------------------------------------------------------------------
__global__ __launch_bounds__(256) void gemm_feat(const __bf16* __restrict__ hs,
                                                 const __bf16* __restrict__ attn,
                                                 const __bf16* __restrict__ Wfb,
                                                 const float* __restrict__ bfv,
                                                 __bf16* __restrict__ featd,
                                                 float* __restrict__ featlast) {
  const int wid = (blockIdx.x * 256 + threadIdx.x) >> 5;  // 2176 tiles
  const int mt = wid / 17, nt = wid % 17;
  const int row0 = mt * 16;
  v8f acc = v8f_zero();
  for (int kt = 0; kt < 16; ++kt) {
    const __bf16* Ap = (kt < 8) ? (hs + (size_t)row0 * DD + kt * 32)
                                : (attn + (size_t)row0 * DD + (kt - 8) * 32);
    acc = wmma_bf16(fragA_bf16(Ap, DD),
                    fragB_bf16(Wfb + (size_t)(nt * 16) * 512 + kt * 32, 512), acc);
  }
  const int lane = threadIdx.x & 31;
  const int col = nt * 16 + (lane & 15);
  const int m0 = row0 + ((lane >> 4) << 3);
  const float bv = (col < 257) ? bfv[col] : 0.f;
#pragma unroll
  for (int r = 0; r < 8; ++r) {
    float v = acc[r] + bv;
    int row = m0 + r;
    if (col < 256) featd[(size_t)row * DD + col] = (__bf16)v;
    else if (col == 256) featlast[row] = v;
  }
}

// ---------------------------------------------------------------------------
// score = feat[:, :256] @ emb^T  + featlast * picked  - (1-emb_mask)*1e20
// picked(b,t,n) == (firstpick[b][n] < t)
// ---------------------------------------------------------------------------
__global__ __launch_bounds__(256) void gemm_score(const __bf16* __restrict__ feat,
                                                  const float* __restrict__ emb,
                                                  const float* __restrict__ emb_mask,
                                                  const float* __restrict__ featlast,
                                                  const int* __restrict__ firstpick,
                                                  float* __restrict__ out) {
  const int b = blockIdx.y;
  const int wid = (blockIdx.x * 256 + threadIdx.x) >> 5;  // 512 tiles/batch
  const int mt = wid >> 7, nt = wid & 127;
  const __bf16* A = feat + (size_t)b * TT * DD;
  const float* Bp = emb + (size_t)b * NN * DD;
  v8f acc = v8f_zero();
  for (int kt = 0; kt < 8; ++kt)
    acc = wmma_bf16(fragA_bf16(A + (size_t)(mt * 16) * DD + kt * 32, DD),
                    fragB_f32(Bp + (size_t)(nt * 16) * DD + kt * 32, DD), acc);
  const int lane = threadIdx.x & 31;
  const int n = nt * 16 + (lane & 15);
  const int m0 = mt * 16 + ((lane >> 4) << 3);
  const float msk = (1.f - emb_mask[(size_t)b * NN + n]) * 1e20f;
  const int fp = firstpick[(size_t)b * NN + n];
#pragma unroll
  for (int r = 0; r < 8; ++r) {
    const int t = m0 + r;
    float v = acc[r] + featlast[b * TT + t] * ((fp < t) ? 1.f : 0.f) - msk;
    out[((size_t)b * TT + t) * NN + n] = v;
  }
}

// ---------------------------------------------------------------------------
extern "C" void kernel_launch(void* const* d_in, const int* in_sizes, int n_in,
                              void* d_out, int out_size, void* d_ws, size_t ws_size,
                              hipStream_t stream) {
  const float* emb      = (const float*)d_in[0];
  const float* emb_mask = (const float*)d_in[1];
  const float* enc      = (const float*)d_in[2];
  const float* enc_mask = (const float*)d_in[3];
  const int*   gt       = (const int*)d_in[4];
  const float* go       = (const float*)d_in[5];
  const float* Wih      = (const float*)d_in[6];
  const float* Whh      = (const float*)d_in[7];
  const float* bih      = (const float*)d_in[8];
  const float* bhh      = (const float*)d_in[9];
  const float* Wdt      = (const float*)d_in[10];
  const float* bdt      = (const float*)d_in[11];
  const float* Wf       = (const float*)d_in[12];
  const float* bfv      = (const float*)d_in[13];
  float* out = (float*)d_out;

  char* wsp = (char*)d_ws;
  auto alloc = [&](size_t bytes) {
    char* p = wsp;
    wsp += (bytes + 255) & ~(size_t)255;
    return p;
  };
  __bf16* Wcat     = (__bf16*)alloc((size_t)G4 * 512 * 2);
  float*  biascat  = (float*) alloc((size_t)G4 * 4);
  __bf16* Wdtb     = (__bf16*)alloc((size_t)DD * DD * 2);
  __bf16* Wfb      = (__bf16*)alloc((size_t)272 * 512 * 2);
  __bf16* Xb       = (__bf16*)alloc((size_t)TT * BB * DD * 2);
  __bf16* hs_bf    = (__bf16*)alloc((size_t)BB * TT * DD * 2);
  __bf16* dec_bf   = (__bf16*)alloc((size_t)BB * TT * DD * 2);
  float*  rawbuf   = (float*) alloc((size_t)BB * TT * SS * 4);
  __bf16* w_bf     = (__bf16*)alloc((size_t)BB * TT * SS * 2);
  __bf16* attn_bf  = (__bf16*)alloc((size_t)BB * TT * DD * 2);
  __bf16* feat_bf  = (__bf16*)alloc((size_t)BB * TT * DD * 2);
  float*  featlast = (float*) alloc((size_t)BB * TT * 4);
  int*    firstpick= (int*)   alloc((size_t)BB * NN * 4);

  prep_kernel<<<512, 256, 0, stream>>>(Wih, Whh, bih, bhh, Wdt, Wf,
                                       Wcat, biascat, Wdtb, Wfb, firstpick);
  fp_fill_kernel<<<8, 256, 0, stream>>>(gt, firstpick);
  build_x_kernel<<<TT * BB, 256, 0, stream>>>(emb, gt, go, Xb);
  lstm_kernel<<<1, 1024, 0, stream>>>(Xb, Wcat, biascat, hs_bf);
  gemm_dec<<<256, 256, 0, stream>>>(hs_bf, Wdtb, bdt, dec_bf);
  gemm_raw<<<dim3(64, BB), 256, 0, stream>>>(dec_bf, enc, enc_mask, rawbuf);
  softmax_kernel<<<BB * TT, 256, 0, stream>>>(rawbuf, w_bf);
  gemm_attn<<<dim3(4, BB), 256, 0, stream>>>(w_bf, enc, attn_bf);
  gemm_feat<<<272, 256, 0, stream>>>(hs_bf, attn_bf, Wfb, bfv, feat_bf, featlast);
  gemm_score<<<dim3(64, BB), 256, 0, stream>>>(feat_bf, emb, emb_mask,
                                               featlast, firstpick, out);
}
